// DynamicGraphBuilder_4492535791884
// MI455X (gfx1250) — matrix operand order repaired and checked by reading, not compiled, not run on hardware
//
#include <hip/hip_runtime.h>
#include <hip/hip_bf16.h>

typedef _Float16 v16h __attribute__((ext_vector_type(16)));
typedef _Float16 v8h  __attribute__((ext_vector_type(8)));
typedef _Float16 v4h  __attribute__((ext_vector_type(4)));
typedef float    v8f  __attribute__((ext_vector_type(8)));

#define DIMS      256
#define KSPATIAL  30
#define KFEAT     6
#define NBUCKETS  1024
#define EQ_CAP    144   // capacity for threshold-bucket ties
#define CARR      (KSPATIAL + EQ_CAP)

// ---------------------------------------------------------------------------
// Kernel 1: F.normalize(ema_feat, dim=-1) -> f16 row (one wave32 per row)
// ---------------------------------------------------------------------------
__global__ __launch_bounds__(32)
void normalize_f16_kernel(const float* __restrict__ feat,
                          _Float16* __restrict__ out, int N) {
    const int i = blockIdx.x;
    const int l = threadIdx.x;                  // 0..31, wave32
    const float* row = feat + (size_t)i * DIMS;

    float4 a0 = ((const float4*)row)[l];        // elems 4l .. 4l+3
    float4 a1 = ((const float4*)row)[l + 32];   // elems 128+4l ..
    float ss = a0.x*a0.x + a0.y*a0.y + a0.z*a0.z + a0.w*a0.w
             + a1.x*a1.x + a1.y*a1.y + a1.z*a1.z + a1.w*a1.w;
    // full-wave butterfly reduction (wave32 -> all lanes hold the sum)
    ss += __shfl_xor(ss, 1);
    ss += __shfl_xor(ss, 2);
    ss += __shfl_xor(ss, 4);
    ss += __shfl_xor(ss, 8);
    ss += __shfl_xor(ss, 16);
    const float scale = 1.0f / fmaxf(sqrtf(ss), 1e-12f);

    _Float16* orow = out + (size_t)i * DIMS;
    v4h o0, o1;
    o0[0] = (_Float16)(a0.x * scale); o0[1] = (_Float16)(a0.y * scale);
    o0[2] = (_Float16)(a0.z * scale); o0[3] = (_Float16)(a0.w * scale);
    o1[0] = (_Float16)(a1.x * scale); o1[1] = (_Float16)(a1.y * scale);
    o1[2] = (_Float16)(a1.z * scale); o1[3] = (_Float16)(a1.w * scale);
    ((v4h*)orow)[l]      = o0;
    ((v4h*)orow)[l + 32] = o1;
}

// ---------------------------------------------------------------------------
// Kernel 2: exact spatial 30-NN per query via LDS histogram threshold select.
// Ordering matches lax.top_k(-d2): ascending d2, ties broken by lower index.
// ---------------------------------------------------------------------------
__global__ __launch_bounds__(256)
void spatial_topk_kernel(const int* __restrict__ pos,
                         int* __restrict__ sp, int N) {
    const int i = blockIdx.x;
    const int t = threadIdx.x;

    __shared__ unsigned int hist[NBUCKETS];
    __shared__ int cd[CARR];
    __shared__ int cj[CARR];
    __shared__ unsigned int cnt_lo, cnt_eq, thr_bucket;

    for (int b = t; b < NBUCKETS; b += 256) hist[b] = 0u;
    if (t == 0) { cnt_lo = 0u; cnt_eq = 0u; }
    __syncthreads();

    const int qx = pos[2 * i];
    const int qy = pos[2 * i + 1];

    // pass 1: histogram of squared distances (bucket width 4)
    for (int j = t; j < N; j += 256) {
        __builtin_prefetch(&pos[2 * (j + 256)], 0, 0);   // global_prefetch_b8
        const int dx = pos[2 * j]     - qx;
        const int dy = pos[2 * j + 1] - qy;
        const int d2 = dx * dx + dy * dy;
        if (j != i) atomicAdd(&hist[min(d2 >> 2, NBUCKETS - 1)], 1u);
    }
    __syncthreads();

    if (t == 0) {
        unsigned int cum = 0u;
        int T = NBUCKETS - 1;
        for (int b = 0; b < NBUCKETS; ++b) {
            cum += hist[b];
            if (cum >= (unsigned)KSPATIAL) { T = b; break; }
        }
        thr_bucket = (unsigned)T;
    }
    __syncthreads();
    const int T = (int)thr_bucket;

    // pass 2: collect candidates strictly below the threshold bucket (<30 of
    // them by construction) and the ties inside the threshold bucket.
    for (int j = t; j < N; j += 256) {
        if (j == i) continue;
        const int dx = pos[2 * j]     - qx;
        const int dy = pos[2 * j + 1] - qy;
        const int d2 = dx * dx + dy * dy;
        const int b  = min(d2 >> 2, NBUCKETS - 1);
        if (b < T) {
            const unsigned s = atomicAdd(&cnt_lo, 1u);   // s < 30 guaranteed
            cd[s] = d2; cj[s] = j;
        } else if (b == T) {
            const unsigned s = atomicAdd(&cnt_eq, 1u);
            if (s < (unsigned)EQ_CAP) { cd[KSPATIAL + s] = d2; cj[KSPATIAL + s] = j; }
        }
    }
    __syncthreads();

    const int nlo  = (int)cnt_lo;
    const int ne   = (int)min(cnt_eq, (unsigned)EQ_CAP);
    const int ntot = nlo + ne;                  // >= 30 always

    // exact rank of each collected entry by (d2, j); ranks are unique.
    if (t < ntot) {
        const int e   = (t < nlo) ? t : (KSPATIAL + (t - nlo));
        const int d2e = cd[e];
        const int je  = cj[e];
        int rank = 0;
        for (int s = 0; s < ntot; ++s) {
            const int si = (s < nlo) ? s : (KSPATIAL + (s - nlo));
            const int ds = cd[si];
            const int js = cj[si];
            rank += (ds < d2e) || (ds == d2e && js < je);
        }
        if (rank < KSPATIAL) sp[(size_t)i * KSPATIAL + rank] = je;
    }
}

// ---------------------------------------------------------------------------
// Kernel 3: cosine sims via v_wmma_f32_16x16x32_f16, top-6 + softmax.
// One wave per query. A = 16 candidate rows, B = query broadcast over N.
// A layout (16-bit A 16x32): lanes 0-15 hold K 0..15,
// lanes 16-31 hold K 16..31 -> two contiguous 16B loads per lane.
// B broadcast: lanes 0-15 hold K 0..15, lanes 16-31 hold K 16..31 (contig 32B).
// ---------------------------------------------------------------------------
__global__ __launch_bounds__(32)
void cosine_topk_kernel(const _Float16* __restrict__ f16,
                        const int* __restrict__ sp,
                        float* __restrict__ out_idx,
                        float* __restrict__ out_w, int N) {
    const int i = blockIdx.x;
    const int l = threadIdx.x;                  // 0..31

    __shared__ float sims[2 * 16];
    __shared__ float top6[KFEAT];
    __shared__ int   cidx[32];

    cidx[l] = sp[(size_t)i * KSPATIAL + ((l < KSPATIAL) ? l : 0)]; // pad 30,31
    __syncthreads();

    const int m = l & 15;
    const _Float16* q  = f16 + (size_t)i * DIMS;
    const _Float16* r0 = f16 + (size_t)cidx[m] * DIMS;        // M-tile 0
    const _Float16* r1 = f16 + (size_t)cidx[16 + m] * DIMS;   // M-tile 1
    const int offA = (l < 16) ? 0 : 16;
    const int offB = (l < 16) ? 0 : 16;

    v8f acc0 = {};
    v8f acc1 = {};
#pragma unroll
    for (int kb = 0; kb < DIMS; kb += 32) {
        const v16h b   = *(const v16h*)(q  + kb + offB);
        const v8h a0lo = *(const v8h*)(r0 + kb + offA);
        const v8h a0hi = *(const v8h*)(r0 + kb + 8 + offA);
        const v8h a1lo = *(const v8h*)(r1 + kb + offA);
        const v8h a1hi = *(const v8h*)(r1 + kb + 8 + offA);
        v16h a0, a1;
#pragma unroll
        for (int e = 0; e < 8; ++e) {
            a0[e] = a0lo[e]; a0[e + 8] = a0hi[e];
            a1[e] = a1lo[e]; a1[e + 8] = a1hi[e];
        }
        acc0 = __builtin_amdgcn_wmma_f32_16x16x32_f16(
                   false, a0, false, b, (short)0, acc0, false, false);
        acc1 = __builtin_amdgcn_wmma_f32_16x16x32_f16(
                   false, a1, false, b, (short)0, acc1, false, false);
    }

    // D layout: lane group 0 VGPR v -> row v; lane group 1 VGPR v -> row v+8.
    if (l == 0) {
#pragma unroll
        for (int v = 0; v < 8; ++v) { sims[v] = acc0[v]; sims[16 + v] = acc1[v]; }
    }
    if (l == 16) {
#pragma unroll
        for (int v = 0; v < 8; ++v) { sims[8 + v] = acc0[v]; sims[24 + v] = acc1[v]; }
    }
    __syncthreads();

    // top-6 of 30 by exact rank (descending sim, ties -> lower index first)
    if (l < KSPATIAL) {
        const float s = sims[l];
        int rank = 0;
        for (int j = 0; j < KSPATIAL; ++j) {
            const float sj = sims[j];
            rank += (sj > s) || (sj == s && j < l);
        }
        if (rank < KFEAT) {
            out_idx[(size_t)i * KFEAT + rank] = (float)cidx[l];
            top6[rank] = s;
        }
    }
    __syncthreads();

    if (l == 0) {
        const float mx = top6[0];                 // sorted descending
        float e[KFEAT];
        float sum = 0.0f;
#pragma unroll
        for (int k = 0; k < KFEAT; ++k) { e[k] = __expf(top6[k] - mx); sum += e[k]; }
        const float inv = 1.0f / sum;
#pragma unroll
        for (int k = 0; k < KFEAT; ++k) out_w[(size_t)i * KFEAT + k] = e[k] * inv;
    }
}

// ---------------------------------------------------------------------------
extern "C" void kernel_launch(void* const* d_in, const int* in_sizes, int n_in,
                              void* d_out, int out_size, void* d_ws, size_t ws_size,
                              hipStream_t stream) {
    const float* feat = (const float*)d_in[0];   // [N, 256] f32
    const int*   pos  = (const int*)d_in[1];     // [N, 2] int
    const int N = in_sizes[0] / DIMS;

    _Float16* f16 = (_Float16*)d_ws;                              // N*256 f16
    int* sp = (int*)((char*)d_ws + (size_t)N * DIMS * sizeof(_Float16)); // N*30

    float* out_idx = (float*)d_out;                 // [N, 6] indices (as f32)
    float* out_w   = (float*)d_out + (size_t)N * KFEAT; // [N, 6] softmax wts

    normalize_f16_kernel<<<N, 32, 0, stream>>>(feat, f16, N);
    spatial_topk_kernel <<<N, 256, 0, stream>>>(pos, sp, N);
    cosine_topk_kernel  <<<N, 32, 0, stream>>>(f16, sp, out_idx, out_w, N);
}
